// model8_27814208209101
// MI455X (gfx1250) — compile-verified, matrix operand order checked by reading, not run on hardware
//
#include <hip/hip_runtime.h>
#include <hip/hip_bf16.h>

typedef __attribute__((ext_vector_type(16))) _Float16 v16h;
typedef __attribute__((ext_vector_type(8)))  float    v8f;

#define HIDDEN   16
#define NN       48
#define DDIM     88
#define K1TILES  6    // K=176 padded to 192
#define N1TILES  3    // N=48
#define K2TILES  2    // K=48 padded to 64
#define N2TILES  6    // N=88 padded to 96
#define WAVES_PB 2    // waves per block
#define TPB      (WAVES_PB * 32)

// Fragment-linear LDS layout helpers (CDNA5 16-bit A/B striping, ISA 7.12.2):
//  A (16x32, MxK): lane = m + 16*((k>>3)&1); half j = (k&7) + 8*((k>>4)&1)
//  B (32x16, KxN): lane = n + 16*((k>>3)&1); half j = (k&7) + 8*((k>>4)&1)
//  inverse: kp = (j&7) + 8*((lane>>4)&1) + 16*(j>>3)

__global__ __launch_bounds__(TPB)
void tones_generator_kernel(
    const int*   __restrict__ w_idx,  const int*   __restrict__ x_idx,
    const float* __restrict__ y,
    const float* __restrict__ W_xh,   const float* __restrict__ b_xh,
    const float* __restrict__ W_wh,   const float* __restrict__ b_wh,
    const float* __restrict__ W_yh,   const float* __restrict__ b_yh,
    const float* __restrict__ W_conv, const float* __restrict__ b_conv,
    const float* __restrict__ W_hl,   const float* __restrict__ b_hl,
    float* __restrict__ out)
{
  __shared__ v16h  sB1[K1TILES * N1TILES * 32];      // 18 KB  W_yh fragments
  __shared__ v16h  sB2[K2TILES * N2TILES * 32];      // 12 KB  W_hl fragments
  __shared__ v16h  sA1[WAVES_PB][K1TILES * 32];      // 12 KB  conv(y) fragments
  __shared__ v16h  sH [WAVES_PB][K2TILES * 32];      //  4 KB  hidden fragments
  __shared__ float sHall[WAVES_PB][16 * NN];         //  6 KB  one-hot + bias term

  const int tid    = threadIdx.x;
  const int lane   = tid & 31;
  const int waveid = tid >> 5;
  const int row0   = (blockIdx.x * WAVES_PB + waveid) * 16;

  // ---- Stage 0a: W_yh -> B fragments.  B1[k][n] = (k<176) ? W_yh[n*176+k] : 0
  {
    _Float16* p = (_Float16*)sB1;
    for (int idx = tid; idx < K1TILES * N1TILES * 32 * 16; idx += TPB) {
      int j    = idx & 15;
      int ln   = (idx >> 4) & 31;
      int frag = idx >> 9;                       // ktile*N1TILES + ntile
      int ktile = frag / N1TILES;
      int ntile = frag % N1TILES;
      int kp = (j & 7) + 8 * ((ln >> 4) & 1) + 16 * (j >> 3);
      int k  = ktile * 32 + kp;
      int n  = ntile * 16 + (ln & 15);
      float v = (k < 176) ? W_yh[n * 176 + k] : 0.0f;
      p[idx] = (_Float16)v;
    }
  }
  // ---- Stage 0b: W_hl -> B fragments.  B2[k][n] = (k<48 && n<88) ? W_hl[n*48+k] : 0
  {
    _Float16* p = (_Float16*)sB2;
    for (int idx = tid; idx < K2TILES * N2TILES * 32 * 16; idx += TPB) {
      int j    = idx & 15;
      int ln   = (idx >> 4) & 31;
      int frag = idx >> 9;
      int ktile = frag / N2TILES;
      int ntile = frag % N2TILES;
      int kp = (j & 7) + 8 * ((ln >> 4) & 1) + 16 * (j >> 3);
      int k  = ktile * 32 + kp;
      int n  = ntile * 16 + (ln & 15);
      float v = (k < NN && n < DDIM) ? W_hl[n * NN + k] : 0.0f;
      p[idx] = (_Float16)v;
    }
  }

  // ---- Stage 1: zero this wave's A1 / H fragment regions (covers K padding)
  {
    unsigned* pa = (unsigned*)sA1[waveid];               // 3072 halves
    for (int i = lane; i < K1TILES * 32 * 16 / 2; i += 32) pa[i] = 0u;
    unsigned* ph = (unsigned*)sH[waveid];                // 1024 halves
    for (int i = lane; i < K2TILES * 32 * 16 / 2; i += 32) ph[i] = 0u;
  }

  // ---- Stage 2: one-hot column gathers + biases (f32, added pre-ReLU)
  for (int idx = lane; idx < 16 * NN; idx += 32) {
    int r = idx / NN, n = idx % NN;
    int gr = row0 + r;
    int xi = x_idx[gr], wi = w_idx[gr];
    sHall[waveid][idx] = W_xh[n * HIDDEN + xi] + W_wh[n * HIDDEN + wi]
                       + b_xh[n] + b_wh[n] + b_yh[n];
  }

  // ---- Stage 3: Conv1d(1->2,k=3,pad=1)+ReLU, scattered into A1 fragments.
  //      lane -> (row r = lane>>1, channel c = lane&1); K index = c*88 + d
  {
    int r = lane >> 1, c = lane & 1;
    const float* yrow = y + (size_t)(row0 + r) * DDIM;
    float wc0 = W_conv[c * 3 + 0], wc1 = W_conv[c * 3 + 1], wc2 = W_conv[c * 3 + 2];
    float bc  = b_conv[c];
    _Float16* pa = (_Float16*)sA1[waveid];
    float ym = 0.0f, yc = yrow[0];
    for (int d = 0; d < DDIM; ++d) {
      float yn = (d < DDIM - 1) ? yrow[d + 1] : 0.0f;
      float v  = fmaxf(bc + wc0 * ym + wc1 * yc + wc2 * yn, 0.0f);
      int k  = c * DDIM + d;
      int kt = k >> 5, kp = k & 31;
      int la = r + 16 * ((kp >> 3) & 1);
      int j  = (kp & 7) + 8 * ((kp >> 4) & 1);
      pa[(kt * 32 + la) * 16 + j] = (_Float16)v;
      ym = yc; yc = yn;
    }
  }

  __syncthreads();

  const int mbase = (lane >> 4) * 8;    // C/D layout: lanes 16-31 hold M=8..15
  const int ncol  = lane & 15;

  // ---- GEMM1: H[16x48] = A1[16x192] * B1[192x48]; +one-hot, ReLU, -> H frags
  for (int ntile = 0; ntile < N1TILES; ++ntile) {
    v8f acc = {};
    #pragma unroll
    for (int ktile = 0; ktile < K1TILES; ++ktile) {
      v16h a = sA1[waveid][ktile * 32 + lane];
      v16h b = sB1[(ktile * N1TILES + ntile) * 32 + lane];
      acc = __builtin_amdgcn_wmma_f32_16x16x32_f16(
          false, a, false, b, (short)0, acc, false, false);
    }
    _Float16* ph = (_Float16*)sH[waveid];
    int n = ntile * 16 + ncol;
    #pragma unroll
    for (int i = 0; i < 8; ++i) {
      int m = mbase + i;
      float v = fmaxf(acc[i] + sHall[waveid][m * NN + n], 0.0f);
      // H feeds GEMM2 as the A-matrix with k = n
      int kt = n >> 5, kp = n & 31;
      int la = m + 16 * ((kp >> 3) & 1);
      int j  = (kp & 7) + 8 * ((kp >> 4) & 1);
      ph[(kt * 32 + la) * 16 + j] = (_Float16)v;
    }
  }

  __syncthreads();

  // ---- GEMM2: out[16x88] = H[16x64] * B2[64x96]; + b_hl, store (n<88 only)
  for (int ntile = 0; ntile < N2TILES; ++ntile) {
    v8f acc = {};
    #pragma unroll
    for (int ktile = 0; ktile < K2TILES; ++ktile) {
      v16h a = sH[waveid][ktile * 32 + lane];
      v16h b = sB2[(ktile * N2TILES + ntile) * 32 + lane];
      acc = __builtin_amdgcn_wmma_f32_16x16x32_f16(
          false, a, false, b, (short)0, acc, false, false);
    }
    int n = ntile * 16 + ncol;
    if (n < DDIM) {
      float bias = b_hl[n];
      #pragma unroll
      for (int i = 0; i < 8; ++i) {
        int m = mbase + i;
        out[(size_t)(row0 + m) * DDIM + n] = acc[i] + bias;
      }
    }
  }
}

extern "C" void kernel_launch(void* const* d_in, const int* in_sizes, int n_in,
                              void* d_out, int out_size, void* d_ws, size_t ws_size,
                              hipStream_t stream) {
  const int*   w_idx  = (const int*)  d_in[0];   // [B,T,1] int32
  const int*   x_idx  = (const int*)  d_in[1];   // [B,T,1] int32
  const float* y      = (const float*)d_in[2];   // [B,T,88]
  const float* W_xh   = (const float*)d_in[3];   // [48,16]
  const float* b_xh   = (const float*)d_in[4];   // [48]
  const float* W_wh   = (const float*)d_in[5];   // [48,16]
  const float* b_wh   = (const float*)d_in[6];   // [48]
  const float* W_yh   = (const float*)d_in[7];   // [48,176]
  const float* b_yh   = (const float*)d_in[8];   // [48]
  const float* W_conv = (const float*)d_in[9];   // [2,1,3]
  const float* b_conv = (const float*)d_in[10];  // [2]
  const float* W_hl   = (const float*)d_in[11];  // [88,48]
  const float* b_hl   = (const float*)d_in[12];  // [88]
  float* out = (float*)d_out;                    // [B,T,88] f32

  const int n_rows = in_sizes[0];                // B*T = 131072 (multiple of 32)
  const int rows_per_block = 16 * WAVES_PB;
  const int blocks = n_rows / rows_per_block;    // 4096

  tones_generator_kernel<<<blocks, TPB, 0, stream>>>(
      w_idx, x_idx, y, W_xh, b_xh, W_wh, b_wh, W_yh, b_yh,
      W_conv, b_conv, W_hl, b_hl, out);
}
